// BertDST_23390391894444
// MI455X (gfx1250) — matrix operand order repaired and checked by reading, not compiled, not run on hardware
//
#include <hip/hip_runtime.h>
#include <hip/hip_bf16.h>

typedef __attribute__((ext_vector_type(16))) _Float16 v16h;
typedef __attribute__((ext_vector_type(8)))  float    v8f;

#define Bdim   64
#define Ldim   512
#define Ddim   768
#define Sdim   30
#define NDOM   5
#define NGATE  3
#define PREDICT_ID 2
#define NPAD   64   // padded GEMM N: [0..29]=W_start, [32..61]=W_end

// flat output offsets (floats), in reference return order
#define OFF_DOM   0
#define OFF_SPP   (Bdim*NDOM)                    // 320
#define OFF_SGP   (OFF_SPP + Bdim*Sdim)          // 2240
#define OFF_PTR   (OFF_SGP + Bdim*Sdim*NGATE)    // 8000
#define OFF_GATE  (OFF_PTR + Bdim*Sdim)          // 9920
#define OFF_SP    (OFF_GATE + Bdim*Sdim)         // 11840
#define OFF_EP    (OFF_SP + Bdim*Sdim*Ldim)      // 994880

// ---------------- K0: pack W_start/W_end into dense f16 B matrix (64 x 768) ----------------
__global__ __launch_bounds__(256) void k0_packB(
    const float* __restrict__ W_start, const float* __restrict__ W_end,
    _Float16* __restrict__ Bpack)
{
    const int idx = blockIdx.x * 256 + threadIdx.x;   // NPAD*Ddim total
    const int n = idx / Ddim, k = idx % Ddim;
    float v = 0.0f;
    if (n < Sdim)                      v = W_start[n * Ddim + k];
    else if (n >= 32 && n < 32 + Sdim) v = W_end[(n - 32) * Ddim + k];
    Bpack[idx] = (_Float16)v;
}

// ---------------- K1: tiny linears + cumsum/scatter maps ----------------
__global__ __launch_bounds__(64) void k1_small(
    const float* __restrict__ cls, const float* __restrict__ W_domain,
    const float* __restrict__ b_domain, const float* __restrict__ W_slot,
    const float* __restrict__ b_slot, const int* __restrict__ slot_pointer,
    const int* __restrict__ slot_gate, float* __restrict__ out,
    int* __restrict__ posmap, int* __restrict__ invmap)
{
    __shared__ float cls_sm[Ddim];
    const int b = blockIdx.x;
    const int t = threadIdx.x;
    for (int i = t; i < Ddim; i += 64) cls_sm[i] = cls[b * Ddim + i];
    if (t < Sdim) {
        posmap[b * Sdim + t] = -1;
        invmap[b * Sdim + t] = -1;
        out[OFF_PTR  + b * Sdim + t] = (float)slot_pointer[b * Sdim + t];
        out[OFF_GATE + b * Sdim + t] = (float)slot_gate[b * Sdim + t];
    }
    __syncthreads();
    if (t < NDOM) {
        float acc = b_domain[t];
        for (int d = 0; d < Ddim; ++d) acc += cls_sm[d] * W_domain[d * NDOM + t];
        out[OFF_DOM + b * NDOM + t] = acc;
    } else if (t < NDOM + Sdim) {
        const int s = t - NDOM;
        float acc = b_slot[s];
        for (int d = 0; d < Ddim; ++d) acc += cls_sm[d] * W_slot[d * Sdim + s];
        out[OFF_SPP + b * Sdim + s] = 1.0f / (1.0f + __expf(-acc));
    }
    if (t == 0) {
        int csum = 0, kc = 0;
        for (int s = 0; s < Sdim; ++s) {
            const int ptr = (slot_pointer[b * Sdim + s] == 1);
            csum += ptr;
            int j = csum - 1;
            if (j < 0) j = 0; if (j > Sdim - 1) j = Sdim - 1;
            if (ptr) {
                posmap[b * Sdim + (csum - 1)] = s;                  // sel row m -> slot s
                if (slot_gate[b * Sdim + j] == PREDICT_ID) {
                    ++kc;
                    invmap[b * Sdim + (kc - 1)] = s;                // predict row k -> slot s
                }
            }
        }
    }
}

// ---------------- K2: G[s,d,g] = sum_e Wg[s,d,e]*W_gate[e,g]; bgW[s,g] ----------------
__global__ __launch_bounds__(256) void k2_fold(
    const float* __restrict__ Wg, const float* __restrict__ bg,
    const float* __restrict__ W_gate, float* __restrict__ G, float* __restrict__ bgW)
{
    __shared__ float wg_sm[Ddim * NGATE];
    const int t = threadIdx.x;
    for (int i = t; i < Ddim * NGATE; i += 256) wg_sm[i] = W_gate[i];
    __syncthreads();
    const int wave = t >> 5, lane = t & 31;
    const int row = blockIdx.x * 8 + wave;
    const int NROW = Sdim * Ddim + Sdim;
    if (row >= NROW) return;
    const float* src = (row < Sdim * Ddim)
        ? (Wg + (size_t)row * Ddim)
        : (bg + (size_t)(row - Sdim * Ddim) * Ddim);
    float a0 = 0.f, a1 = 0.f, a2 = 0.f;
    for (int e0 = lane * 4; e0 < Ddim; e0 += 128) {
        const float4 v = *(const float4*)(src + e0);
        a0 += v.x*wg_sm[(e0+0)*3+0] + v.y*wg_sm[(e0+1)*3+0] + v.z*wg_sm[(e0+2)*3+0] + v.w*wg_sm[(e0+3)*3+0];
        a1 += v.x*wg_sm[(e0+0)*3+1] + v.y*wg_sm[(e0+1)*3+1] + v.z*wg_sm[(e0+2)*3+1] + v.w*wg_sm[(e0+3)*3+1];
        a2 += v.x*wg_sm[(e0+0)*3+2] + v.y*wg_sm[(e0+1)*3+2] + v.z*wg_sm[(e0+2)*3+2] + v.w*wg_sm[(e0+3)*3+2];
    }
    for (int off = 16; off > 0; off >>= 1) {
        a0 += __shfl_down(a0, off, 32);
        a1 += __shfl_down(a1, off, 32);
        a2 += __shfl_down(a2, off, 32);
    }
    if (lane == 0) {
        if (row < Sdim * Ddim) { G[(size_t)row*3+0]=a0; G[(size_t)row*3+1]=a1; G[(size_t)row*3+2]=a2; }
        else { const int s = row - Sdim*Ddim; bgW[s*3+0]=a0; bgW[s*3+1]=a1; bgW[s*3+2]=a2; }
    }
}

// ---------------- K3: slot_gate_prob via folded G ----------------
__global__ __launch_bounds__(256) void k3_gate(
    const float* __restrict__ cls, const float* __restrict__ G,
    const float* __restrict__ bgW, const float* __restrict__ b_gate,
    const int* __restrict__ posmap, float* __restrict__ out)
{
    const int t = threadIdx.x, wave = t >> 5, lane = t & 31;
    const int gw = blockIdx.x * 8 + wave;
    if (gw >= Bdim * Sdim) return;
    const int b = gw / Sdim, m = gw % Sdim;
    const int s = posmap[b * Sdim + m];
    float sc0, sc1, sc2;
    if (s < 0) {
        sc0 = b_gate[0]; sc1 = b_gate[1]; sc2 = b_gate[2];   // zero sel row
    } else {
        float a0 = 0.f, a1 = 0.f, a2 = 0.f;
        for (int d = lane; d < Ddim; d += 32) {
            const float c = cls[b * Ddim + d];
            const float* g = G + (size_t)(s * Ddim + d) * 3;
            a0 += c * g[0]; a1 += c * g[1]; a2 += c * g[2];
        }
        for (int off = 16; off > 0; off >>= 1) {
            a0 += __shfl_down(a0, off, 32);
            a1 += __shfl_down(a1, off, 32);
            a2 += __shfl_down(a2, off, 32);
        }
        sc0 = a0 + bgW[s*3+0] + b_gate[0];
        sc1 = a1 + bgW[s*3+1] + b_gate[1];
        sc2 = a2 + bgW[s*3+2] + b_gate[2];
    }
    if (lane == 0) {
        const float mx = fmaxf(sc0, fmaxf(sc1, sc2));
        const float e0 = __expf(sc0 - mx), e1 = __expf(sc1 - mx), e2 = __expf(sc2 - mx);
        const float inv = 1.0f / (e0 + e1 + e2);
        float* o = out + OFF_SGP + (size_t)(b * Sdim + m) * 3;
        o[0] = e0 * inv; o[1] = e1 * inv; o[2] = e2 * inv;
    }
}

// ---------------- K4: WMMA GEMM  C[m,n] = H[m,:] . Bpack[n,:]  (sync-free, no LDS) ----------------
__global__ __launch_bounds__(256) void k4_gemm(
    const float* __restrict__ H, const _Float16* __restrict__ Bpack,
    const float* __restrict__ b_start, const float* __restrict__ b_end,
    float* __restrict__ scoreS, float* __restrict__ scoreE)
{
    const int t = threadIdx.x, wave = t >> 5, lane = t & 31;
    const int mrow = blockIdx.x * 128 + wave * 16;     // 8 waves x 16 rows
    const int mA = lane & 15;
    const int hi = lane >> 4;                          // lane group 0/1
    const float* hrow = H + (size_t)(mrow + mA) * Ddim;
    // lane's B column base: col = nt*16 + mA; per-lane contiguous 16-half run at k0 + hi*16
    const _Float16* bbase = Bpack + (size_t)mA * Ddim + hi * 16;

    v8f zero = {};
    v8f acc[4] = { zero, zero, zero, zero };

    for (int k0 = 0; k0 < Ddim; k0 += 32) {
        // A fragment: one contiguous 16-k run at k0 + hi*16 (matches B)
        const int c0 = k0 + hi * 16;
        const float4 v0 = *(const float4*)(hrow + c0);
        const float4 v1 = *(const float4*)(hrow + c0 + 4);
        const float4 v2 = *(const float4*)(hrow + c0 + 8);
        const float4 v3 = *(const float4*)(hrow + c0 + 12);
        v16h a;
        a[0]=(_Float16)v0.x;  a[1]=(_Float16)v0.y;  a[2]=(_Float16)v0.z;  a[3]=(_Float16)v0.w;
        a[4]=(_Float16)v1.x;  a[5]=(_Float16)v1.y;  a[6]=(_Float16)v1.z;  a[7]=(_Float16)v1.w;
        a[8]=(_Float16)v2.x;  a[9]=(_Float16)v2.y;  a[10]=(_Float16)v2.z; a[11]=(_Float16)v2.w;
        a[12]=(_Float16)v3.x; a[13]=(_Float16)v3.y; a[14]=(_Float16)v3.z; a[15]=(_Float16)v3.w;

        #pragma unroll
        for (int nt = 0; nt < 4; ++nt) {
            const v16h bf = *(const v16h*)(bbase + (size_t)nt * 16 * Ddim + k0);
            acc[nt] = __builtin_amdgcn_wmma_f32_16x16x32_f16(
                false, a, false, bf, (short)0, acc[nt], false, false);
        }
    }

    // epilogue: C layout — VGPR r: lanes 0-15 M=r, lanes 16-31 M=r+8; N=lane&15
    const int col0 = lane & 15;
    const int mofs = hi * 8;
    #pragma unroll
    for (int nt = 0; nt < 4; ++nt) {
        const int col = nt * 16 + col0;
        const bool isS = (col < Sdim);
        const bool isE = (col >= 32 && col < 32 + Sdim);
        if (!isS && !isE) continue;
        const int s = isS ? col : (col - 32);
        const float bias = isS ? b_start[s] : b_end[s];
        float* base = isS ? scoreS : scoreE;
        #pragma unroll
        for (int r = 0; r < 8; ++r) {
            const int m = mrow + mofs + r;
            const int b = m >> 9;            // / 512
            const int l = m & 511;
            base[((size_t)b * Sdim + s) * Ldim + l] = acc[nt][r] + bias;
        }
    }
}

// ---------------- K5: scatter-select + softmax over L ----------------
__global__ __launch_bounds__(256) void k5_softmax(
    const float* __restrict__ scoreS, const float* __restrict__ scoreE,
    const int* __restrict__ invmap, float* __restrict__ out)
{
    __shared__ float red[256];
    const int idx = blockIdx.x;
    const int which = idx & 1;
    const int bk = idx >> 1;
    const int b = bk / Sdim, k = bk % Sdim;
    const int t = threadIdx.x;
    float* dst = out + (which ? OFF_EP : OFF_SP) + ((size_t)b * Sdim + k) * Ldim;
    const int s = invmap[b * Sdim + k];
    if (s < 0) {                               // zero row -> uniform softmax
        const float u = 1.0f / (float)Ldim;
        dst[t] = u; dst[t + 256] = u;
        return;
    }
    const float* src = (which ? scoreE : scoreS) + ((size_t)b * Sdim + s) * Ldim;
    const float x0 = src[t], x1 = src[t + 256];
    red[t] = fmaxf(x0, x1);
    __syncthreads();
    for (int off = 128; off > 0; off >>= 1) { if (t < off) red[t] = fmaxf(red[t], red[t + off]); __syncthreads(); }
    const float mx = red[0];
    __syncthreads();
    const float e0 = __expf(x0 - mx), e1 = __expf(x1 - mx);
    red[t] = e0 + e1;
    __syncthreads();
    for (int off = 128; off > 0; off >>= 1) { if (t < off) red[t] += red[t + off]; __syncthreads(); }
    const float inv = 1.0f / red[0];
    dst[t] = e0 * inv; dst[t + 256] = e1 * inv;
}

extern "C" void kernel_launch(void* const* d_in, const int* in_sizes, int n_in,
                              void* d_out, int out_size, void* d_ws, size_t ws_size,
                              hipStream_t stream)
{
    const float* H            = (const float*)d_in[0];
    const float* cls          = (const float*)d_in[1];
    const float* W_domain     = (const float*)d_in[2];
    const float* b_domain     = (const float*)d_in[3];
    const float* W_slot       = (const float*)d_in[4];
    const float* b_slot       = (const float*)d_in[5];
    const float* Wg           = (const float*)d_in[6];
    const float* bg           = (const float*)d_in[7];
    const float* W_gate       = (const float*)d_in[8];
    const float* b_gate       = (const float*)d_in[9];
    const float* W_start      = (const float*)d_in[10];
    const float* b_start      = (const float*)d_in[11];
    const float* W_end        = (const float*)d_in[12];
    const float* b_end        = (const float*)d_in[13];
    const int*   slot_pointer = (const int*)d_in[14];
    const int*   slot_gate    = (const int*)d_in[15];
    float* out = (float*)d_out;

    float* ws     = (float*)d_ws;
    float* G      = ws;                                   // S*D*3
    float* bgW    = G + (size_t)Sdim * Ddim * NGATE;      // S*3
    float* scoreS = bgW + Sdim * NGATE;                   // B*S*L
    float* scoreE = scoreS + (size_t)Bdim * Sdim * Ldim;  // B*S*L
    int*   posmap = (int*)(scoreE + (size_t)Bdim * Sdim * Ldim); // B*S
    int*   invmap = posmap + Bdim * Sdim;                 // B*S
    _Float16* Bpack = (_Float16*)(invmap + Bdim * Sdim);  // NPAD*Ddim f16

    k0_packB<<<(NPAD * Ddim) / 256, 256, 0, stream>>>(W_start, W_end, Bpack);
    k1_small<<<Bdim, 64, 0, stream>>>(cls, W_domain, b_domain, W_slot, b_slot,
                                      slot_pointer, slot_gate, out, posmap, invmap);
    const int nrow = Sdim * Ddim + Sdim;
    k2_fold<<<(nrow + 7) / 8, 256, 0, stream>>>(Wg, bg, W_gate, G, bgW);
    k3_gate<<<(Bdim * Sdim + 7) / 8, 256, 0, stream>>>(cls, G, bgW, b_gate, posmap, out);
    k4_gemm<<<(Bdim * Ldim) / 128, 256, 0, stream>>>(H, Bpack, b_start, b_end,
                                                     scoreS, scoreE);
    k5_softmax<<<Bdim * Sdim * 2, 256, 0, stream>>>(scoreS, scoreE, invmap, out);
}
